// QuaternionPairPhasorBlock_84825604096580
// MI455X (gfx1250) — matrix operand order repaired and checked by reading, not compiled, not run on hardware
//
#include <hip/hip_runtime.h>
#include <math.h>

// Problem constants (from reference): B=4, L=4096, D=1024, K=D/4
#define Bb     4
#define Ll     4096
#define Dd     1024
#define Kq     (Dd / 4)        // 256 quaternions per token
#define MROWS  (Bb * Ll)       // 16384 tokens
#define CHUNKS 16
#define CLEN   (Ll / CHUNKS)   // 256

// GEMM tiling
#define TILE_M 128
#define TILE_N 128
#define KB     32
#define LSTR   40              // padded LDS stride in bf16 elems (80B, 16B-aligned)

typedef __attribute__((ext_vector_type(16))) __bf16 v16bf;
typedef __attribute__((ext_vector_type(8)))  __bf16 v8bf;
typedef __attribute__((ext_vector_type(4)))  __bf16 v4bf;
typedef __attribute__((ext_vector_type(8)))  float  v8f;

// ---------------------------------------------------------------------------
// Tiled bf16 WMMA GEMM: Out[M,N] = epilogue(A[M,Kd] @ W[Kd,N] + bias)
// mode 0: +bias   mode 1: gelu(+bias)   mode 2: +bias +Res (residual)
// ---------------------------------------------------------------------------
__global__ __launch_bounds__(256) void gemm_bf16_wmma(
    const float* __restrict__ A, const float* __restrict__ W,
    const float* __restrict__ bias, float* __restrict__ Out,
    const float* __restrict__ Res, int M, int N, int Kd, int mode)
{
    __shared__ __align__(16) __bf16 As[TILE_M * LSTR];
    __shared__ __align__(16) __bf16 Bs[TILE_N * LSTR];

    const int tid  = threadIdx.x;
    const int wave = tid >> 5;
    const int lane = tid & 31;
    const int half = lane >> 4;
    const int mr   = lane & 15;
    const int wm   = wave & 3;     // 4 waves along M (32-row bands)
    const int wn   = wave >> 2;    // 2 waves along N (64-col bands)
    const int tileM = blockIdx.y * TILE_M;
    const int tileN = blockIdx.x * TILE_N;

    // staging decomposition (256 threads)
    const int ar  = tid >> 3;        // A: 0..31 row group
    const int ac  = (tid & 7) * 4;   // A: col within 32-wide k-slab
    const int bn  = tid & 127;       // B: column owned by this thread
    const int bkh = tid >> 7;        // B: k-half (0 -> K0..15, 1 -> K16..31)

    float4 areg[4];
    float  breg[16];

    auto fetch = [&](int kt) {
        const int k0 = kt * KB;
#pragma unroll
        for (int i = 0; i < 4; ++i)
            areg[i] = *(const float4*)(A + (size_t)(tileM + ar + 32 * i) * Kd + k0 + ac);
        // B: 16 k-strided b32 loads; lanes cover 32 consecutive columns ->
        // each load instruction is a fully coalesced 128B transaction.
        const float* wp = W + (size_t)(k0 + bkh * 16) * N + tileN + bn;
#pragma unroll
        for (int j = 0; j < 16; ++j) breg[j] = wp[(size_t)j * N];
        if (k0 + 2 * KB < Kd) {  // L2 prefetch of the slab after next
            __builtin_prefetch(A + (size_t)(tileM + ar) * Kd + k0 + 2 * KB + ac, 0, 1);
            __builtin_prefetch(W + (size_t)(k0 + 2 * KB + bkh * 16) * N + tileN + bn, 0, 1);
        }
    };

    auto stage = [&]() {
#pragma unroll
        for (int i = 0; i < 4; ++i) {
            v4bf pa = { (__bf16)areg[i].x, (__bf16)areg[i].y,
                        (__bf16)areg[i].z, (__bf16)areg[i].w };
            *(v4bf*)(&As[(ar + 32 * i) * LSTR + ac]) = pa;
        }
        // B stored K-contiguous per column: two aligned b128 LDS stores
        v8bf b0 = { (__bf16)breg[0],  (__bf16)breg[1],  (__bf16)breg[2],  (__bf16)breg[3],
                    (__bf16)breg[4],  (__bf16)breg[5],  (__bf16)breg[6],  (__bf16)breg[7] };
        v8bf b1 = { (__bf16)breg[8],  (__bf16)breg[9],  (__bf16)breg[10], (__bf16)breg[11],
                    (__bf16)breg[12], (__bf16)breg[13], (__bf16)breg[14], (__bf16)breg[15] };
        __bf16* bp = &Bs[bn * LSTR + bkh * 16];
        *(v8bf*)(bp)     = b0;
        *(v8bf*)(bp + 8) = b1;
    };

    v8f acc[2][4];
#pragma unroll
    for (int r = 0; r < 2; ++r)
#pragma unroll
        for (int c = 0; c < 4; ++c)
#pragma unroll
            for (int i = 0; i < 8; ++i) acc[r][c][i] = 0.0f;

    const int nk = Kd / KB;
    fetch(0);
    for (int kt = 0; kt < nk; ++kt) {
        stage();
        __syncthreads();
        if (kt + 1 < nk) fetch(kt + 1);   // overlap HBM with WMMA below

        v16bf af[2], bf[4];
#pragma unroll
        for (int r = 0; r < 2; ++r) {
            // A 16x32 bf16 fragment: lanes0-15 K0-15, lanes16-31 K16-31 (matches B)
            const __bf16* ap = &As[(wm * 32 + r * 16 + mr) * LSTR];
            v8bf lo = *(const v8bf*)(ap + half * 16);
            v8bf hi = *(const v8bf*)(ap + half * 16 + 8);
            af[r] = __builtin_shufflevector(lo, hi, 0,1,2,3,4,5,6,7,8,9,10,11,12,13,14,15);
        }
#pragma unroll
        for (int c = 0; c < 4; ++c) {
            // B 32x16 bf16 fragment: lane = col, lanes0-15 K0-15, lanes16-31 K16-31
            const __bf16* bp = &Bs[(wn * 64 + c * 16 + mr) * LSTR + half * 16];
            v8bf lo = *(const v8bf*)(bp);
            v8bf hi = *(const v8bf*)(bp + 8);
            bf[c] = __builtin_shufflevector(lo, hi, 0,1,2,3,4,5,6,7,8,9,10,11,12,13,14,15);
        }
#pragma unroll
        for (int r = 0; r < 2; ++r)
#pragma unroll
            for (int c = 0; c < 4; ++c)
                acc[r][c] = __builtin_amdgcn_wmma_f32_16x16x32_bf16(
                    false, af[r], false, bf[c], (short)0, acc[r][c], false, false);
        __syncthreads();
    }

    // Epilogue. C/D layout: vgpr i, lane -> row = half*8 + i, col = lane&15.
#pragma unroll
    for (int r = 0; r < 2; ++r) {
        const int row0 = tileM + wm * 32 + r * 16 + half * 8;
#pragma unroll
        for (int c = 0; c < 4; ++c) {
            const int col = tileN + wn * 64 + c * 16 + mr;
            const float bc = bias[col];
#pragma unroll
            for (int i = 0; i < 8; ++i) {
                float v = acc[r][c][i] + bc;
                if (mode == 1) v = 0.5f * v * (1.0f + erff(v * 0.70710678118654752f));
                const size_t idx = (size_t)(row0 + i) * N + col;
                if (mode == 2) v += Res[idx];
                Out[idx] = v;
            }
        }
    }
}

// ---------------------------------------------------------------------------
// Quaternion helpers (component 0 = w)
// ---------------------------------------------------------------------------
__device__ __forceinline__ float4 qmul(float4 a, float4 b) {
    return make_float4(
        a.x * b.x - a.y * b.y - a.z * b.z - a.w * b.w,
        a.x * b.y + a.y * b.x + a.z * b.w - a.w * b.z,
        a.x * b.z - a.y * b.w + a.z * b.x + a.w * b.y,
        a.x * b.w + a.y * b.z - a.z * b.y + a.w * b.x);
}
__device__ __forceinline__ float4 qconj(float4 q) {
    return make_float4(q.x, -q.y, -q.z, -q.w);
}
__device__ __forceinline__ float4 qnormalize(float4 q) {
    float n = sqrtf(q.x * q.x + q.y * q.y + q.z * q.z + q.w * q.w);
    float inv = 1.0f / fmaxf(n, 1e-12f);
    return make_float4(q.x * inv, q.y * inv, q.z * inv, q.w * inv);
}

// normalize key/query; rotated = k * v * conj(k) written over value buffer
__global__ void quat_prepare(float4* __restrict__ kq, float4* __restrict__ qq,
                             float4* __restrict__ val, int n)
{
    int idx = blockIdx.x * blockDim.x + threadIdx.x;
    if (idx >= n) return;
    float4 k = qnormalize(kq[idx]);
    float4 q = qnormalize(qq[idx]);
    qq[idx] = q;                       // keep normalized query for retrieval
    float4 v = val[idx];
    val[idx] = qmul(qmul(k, v), qconj(k));
}

// pass 1: per-chunk channel sums
__global__ void chunk_sums(const float* __restrict__ rot, float* __restrict__ csum)
{
    int t = blockIdx.x * blockDim.x + threadIdx.x;   // B*CHUNKS*D threads
    int d = t % Dd;
    int bc = t / Dd;
    int ch = bc % CHUNKS;
    int b  = bc / CHUNKS;
    const float* p = rot + ((size_t)b * Ll + (size_t)ch * CLEN) * Dd + d;
    float s = 0.0f;
    for (int i = 0; i < CLEN; ++i) s += p[(size_t)i * Dd];
    csum[((size_t)b * CHUNKS + ch) * Dd + d] = s;
}

// pass 2: exclusive scan of the 16 chunk sums per channel
__global__ void scan_chunks(float* __restrict__ csum)
{
    int t = blockIdx.x * blockDim.x + threadIdx.x;   // B*D threads
    int d = t % Dd;
    int b = t / Dd;
    float run = 0.0f;
    for (int ch = 0; ch < CHUNKS; ++ch) {
        size_t i = ((size_t)b * CHUNKS + ch) * Dd + d;
        float v = csum[i];
        csum[i] = run;
        run += v;
    }
}

// pass 3 fused with retrieval:
// memory = offset + local cumsum; ret = conj(q) * memory * q * rsqrt(l+1)
__global__ void retrieve_scan(const float4* __restrict__ rot,
                              const float4* __restrict__ csum,
                              const float4* __restrict__ qq,
                              float4* __restrict__ outr)
{
    int t = blockIdx.x * blockDim.x + threadIdx.x;   // B*CHUNKS*K threads
    int k  = t % Kq;
    int bc = t / Kq;
    int ch = bc % CHUNKS;
    int b  = bc / CHUNKS;
    float4 run = csum[((size_t)b * CHUNKS + ch) * Kq + k];
    for (int i = 0; i < CLEN; ++i) {
        int l = ch * CLEN + i;
        size_t idx = ((size_t)b * Ll + l) * Kq + k;
        float4 r = rot[idx];
        run.x += r.x; run.y += r.y; run.z += r.z; run.w += r.w;
        float4 q = qq[idx];
        float4 ret = qmul(qmul(qconj(q), run), q);
        float inv = rsqrtf((float)(l + 1));
        ret.x *= inv; ret.y *= inv; ret.z *= inv; ret.w *= inv;
        outr[idx] = ret;
    }
}

// per-token layernorm over D=1024 (256 threads x float4)
__global__ __launch_bounds__(256) void layernorm_tok(
    const float* __restrict__ in, const float* __restrict__ gamma,
    const float* __restrict__ beta, float* __restrict__ out)
{
    __shared__ float s1[256], s2[256];
    const int tid = threadIdx.x;
    const size_t base = (size_t)blockIdx.x * Dd;
    float4 v = *(const float4*)(in + base + tid * 4);
    float sum = v.x + v.y + v.z + v.w;
    float sq  = v.x * v.x + v.y * v.y + v.z * v.z + v.w * v.w;
    s1[tid] = sum; s2[tid] = sq;
    __syncthreads();
    for (int off = 128; off > 0; off >>= 1) {
        if (tid < off) { s1[tid] += s1[tid + off]; s2[tid] += s2[tid + off]; }
        __syncthreads();
    }
    const float mean = s1[0] * (1.0f / Dd);
    const float var  = s2[0] * (1.0f / Dd) - mean * mean;
    const float rstd = rsqrtf(var + 1e-5f);
    float4 g  = *(const float4*)(gamma + tid * 4);
    float4 be = *(const float4*)(beta + tid * 4);
    float4 o;
    o.x = (v.x - mean) * rstd * g.x + be.x;
    o.y = (v.y - mean) * rstd * g.y + be.y;
    o.z = (v.z - mean) * rstd * g.z + be.z;
    o.w = (v.w - mean) * rstd * g.w + be.w;
    *(float4*)(out + base + tid * 4) = o;
}

// ---------------------------------------------------------------------------
extern "C" void kernel_launch(void* const* d_in, const int* in_sizes, int n_in,
                              void* d_out, int out_size, void* d_ws, size_t ws_size,
                              hipStream_t stream)
{
    (void)in_sizes; (void)n_in; (void)out_size; (void)ws_size;
    const float* x     = (const float*)d_in[0];
    const float* Wk1   = (const float*)d_in[1];
    const float* bk1   = (const float*)d_in[2];
    const float* Wk2   = (const float*)d_in[3];
    const float* bk2   = (const float*)d_in[4];
    const float* Wq1   = (const float*)d_in[5];
    const float* bq1   = (const float*)d_in[6];
    const float* Wq2   = (const float*)d_in[7];
    const float* bq2   = (const float*)d_in[8];
    const float* Wv    = (const float*)d_in[9];
    const float* bv    = (const float*)d_in[10];
    const float* gamma = (const float*)d_in[11];
    const float* beta  = (const float*)d_in[12];
    const float* Wo    = (const float*)d_in[13];
    const float* bo    = (const float*)d_in[14];
    float* out = (float*)d_out;

    const size_t MD = (size_t)MROWS * Dd;
    float* buf0 = (float*)d_ws;      // key_q raw -> retrieved
    float* buf1 = buf0 + MD;         // query_q raw -> normalized q -> ln out
    float* buf2 = buf1 + MD;         // value -> rotated (scan input, in place)
    float* buf3 = buf2 + MD;         // MLP hidden (reused k then q)
    float* csum = buf3 + MD;         // chunk sums: B*CHUNKS*D

    dim3 gg(Dd / TILE_N, MROWS / TILE_M);   // (8, 128)

    // key MLP
    gemm_bf16_wmma<<<gg, 256, 0, stream>>>(x,    Wk1, bk1, buf3, nullptr, MROWS, Dd, Dd, 1);
    gemm_bf16_wmma<<<gg, 256, 0, stream>>>(buf3, Wk2, bk2, buf0, nullptr, MROWS, Dd, Dd, 0);
    // query MLP
    gemm_bf16_wmma<<<gg, 256, 0, stream>>>(x,    Wq1, bq1, buf3, nullptr, MROWS, Dd, Dd, 1);
    gemm_bf16_wmma<<<gg, 256, 0, stream>>>(buf3, Wq2, bq2, buf1, nullptr, MROWS, Dd, Dd, 0);
    // value projection
    gemm_bf16_wmma<<<gg, 256, 0, stream>>>(x,    Wv,  bv,  buf2, nullptr, MROWS, Dd, Dd, 0);

    const int nq = MROWS * Kq;
    quat_prepare<<<nq / 256, 256, 0, stream>>>(
        (float4*)buf0, (float4*)buf1, (float4*)buf2, nq);

    chunk_sums<<<(Bb * CHUNKS * Dd) / 256, 256, 0, stream>>>(buf2, csum);
    scan_chunks<<<(Bb * Dd) / 256, 256, 0, stream>>>(csum);
    retrieve_scan<<<(Bb * CHUNKS * Kq) / 256, 256, 0, stream>>>(
        (const float4*)buf2, (const float4*)csum, (const float4*)buf1, (float4*)buf0);

    layernorm_tok<<<MROWS, 256, 0, stream>>>(buf0, gamma, beta, buf1);

    // out = x + ln @ Wo + bo
    gemm_bf16_wmma<<<gg, 256, 0, stream>>>(buf1, Wo, bo, out, x, MROWS, Dd, Dd, 2);
}
